// FCNN_3118146257417
// MI455X (gfx1250) — compile-verified, matrix operand order checked by reading, not run on hardware
//
#include <hip/hip_runtime.h>
#include <hip/hip_bf16.h>

// ---------------------------------------------------------------------------
// Sizes (fixed by the reference problem)
// ---------------------------------------------------------------------------
#define N_ATOMS 100000
#define IN_F    1008
#define H1      256
#define H2      192
#define H3      160
#define N_TYPES 5
#define MTILE   64        // atoms per block (4 WMMA M-tiles)
#define MAXT    1572      // >= sum_t ceil(cnt_t/64) worst case (1567)

typedef __attribute__((ext_vector_type(16))) __bf16 v16bf;
typedef __attribute__((ext_vector_type(8)))  __bf16 v8bf;
typedef __attribute__((ext_vector_type(4)))  __bf16 v4bf;
typedef __attribute__((ext_vector_type(8)))  float  v8f;
typedef __attribute__((ext_vector_type(4)))  float  v4f;

// ---------------------------------------------------------------------------
// Workspace layout (bytes)
// hdr ints: [0..4]=counts [5..9]=offsets [10..14]=cursors [15]=numTiles
// ---------------------------------------------------------------------------
static const size_t OFF_HDR  = 0;                                    // 128 B
static const size_t OFF_PERM = 128;                                  // N_ATOMS*4
static const size_t OFF_TILT = OFF_PERM + (size_t)N_ATOMS * 4;       // MAXT*4
static const size_t OFF_TILB = OFF_TILT + (size_t)MAXT * 4;          // MAXT*4
static const size_t OFF_PW1  = (OFF_TILB + (size_t)MAXT * 4 + 63) & ~(size_t)63;
static const size_t SZ_PW1   = (size_t)N_TYPES * 16 * 32 * 512 * 2;  // K pad 1024
static const size_t OFF_PW2  = OFF_PW1 + SZ_PW1;
static const size_t SZ_PW2   = (size_t)N_TYPES * 12 * 8 * 512 * 2;
static const size_t OFF_PW3  = OFF_PW2 + SZ_PW2;
static const size_t SZ_PW3   = (size_t)N_TYPES * 10 * 6 * 512 * 2;

// ---------------------------------------------------------------------------
// bf16 helpers (explicit RNE bit manipulation)
// ---------------------------------------------------------------------------
__device__ __forceinline__ __bf16 f2bf(float f) {
    unsigned u = __builtin_bit_cast(unsigned, f);
    unsigned r = (u + 0x7FFFu + ((u >> 16) & 1u)) >> 16;
    unsigned short s = (unsigned short)r;
    return __builtin_bit_cast(__bf16, s);
}
__device__ __forceinline__ float bf2f(__bf16 b) {
    unsigned short s = __builtin_bit_cast(unsigned short, b);
    unsigned u = ((unsigned)s) << 16;
    return __builtin_bit_cast(float, u);
}
__device__ __forceinline__ int ztype(int z) {
    return z == 1 ? 0 : (z == 6 ? 1 : (z == 7 ? 2 : (z == 8 ? 3 : 4)));
}
__device__ __forceinline__ float celu01(float x) {
    return fmaxf(x, 0.f) + fminf(0.f, 0.1f * (__expf(x * 10.f) - 1.f));
}

// ---------------------------------------------------------------------------
// Prep kernels
// ---------------------------------------------------------------------------
__global__ void k_zero(int* hdr, float* out, int n) {
    int i = blockIdx.x * blockDim.x + threadIdx.x;
    if (i < 32) hdr[i] = 0;
    if (i < n)  out[i] = 0.f;
}

__global__ void k_count(const int* __restrict__ z, int* hdr) {
    int i = blockIdx.x * blockDim.x + threadIdx.x;
    if (i < N_ATOMS) atomicAdd(&hdr[ztype(z[i])], 1);
}

__global__ void k_scan(int* hdr, int* tileT, int* tileB) {
    if (blockIdx.x == 0 && threadIdx.x == 0) {
        int off = 0, nt = 0;
        for (int t = 0; t < N_TYPES; ++t) {
            hdr[5 + t] = off;
            int c = hdr[t];
            for (int i = 0; i < c; i += MTILE) { tileT[nt] = t; tileB[nt] = off + i; ++nt; }
            off += c;
        }
        hdr[15] = nt;
    }
}

__global__ void k_scatter(const int* __restrict__ z, int* hdr, int* perm) {
    int i = blockIdx.x * blockDim.x + threadIdx.x;
    if (i < N_ATOMS) {
        int t = ztype(z[i]);
        int pos = hdr[5 + t] + atomicAdd(&hdr[10 + t], 1);
        perm[pos] = i;
    }
}

// Pack f32 weights [T][K][N] into bf16 WMMA B-fragment layout:
// halves index = ((((t*NT + nt)*NKB + kb)*32 + lane)*16 + e)
// lane: col = lane&15, khalf = lane>>4 ; e -> krel = (e<8 ? e : e+8) + 8*khalf
__global__ void k_pack(const float* __restrict__ W, __bf16* __restrict__ P,
                       int K, int N, int NT, int NKB) {
    long long idx = (long long)blockIdx.x * blockDim.x + threadIdx.x;
    long long total = (long long)N_TYPES * NT * NKB * 512;
    if (idx >= total) return;
    int e    = (int)(idx & 15);
    int lane = (int)((idx >> 4) & 31);
    long long blk = idx >> 9;
    int kb = (int)(blk % NKB); blk /= NKB;
    int nt = (int)(blk % NT);
    int t  = (int)(blk / NT);
    int col   = lane & 15;
    int khalf = lane >> 4;
    int krel  = (e < 8 ? e : e + 8) + 8 * khalf;
    int k = kb * 32 + krel;
    int n = nt * 16 + col;
    float v = (k < K && n < N) ? W[((long long)t * K + k) * N + n] : 0.f;
    P[idx] = f2bf(v);
}

// ---------------------------------------------------------------------------
// WMMA fragment loads
// A (16xK row-major bf16 in LDS): lane row = lane&15, koff = (lane>>4)*8
// e 0..7  <- K = k0 + koff + 0..7 ; e 8..15 <- K = k0 + 16 + koff + 0..7
// ---------------------------------------------------------------------------
__device__ __forceinline__ v16bf load_a_frag(const __bf16* A, int ldK, int k0, int lane) {
    int row  = lane & 15;
    int koff = (lane >> 4) * 8;
    const v8bf* p0 = (const v8bf*)(A + row * ldK + k0 + koff);
    const v8bf* p1 = (const v8bf*)(A + row * ldK + k0 + 16 + koff);
    v8bf lo = *p0, hi = *p1;
    v16bf a;
#pragma unroll
    for (int i = 0; i < 8; ++i) { a[i] = lo[i]; a[i + 8] = hi[i]; }
    return a;
}

__device__ __forceinline__ v16bf load_b_frag(const __bf16* __restrict__ PW,
                                             int t, int NT, int NKB, int n, int kb,
                                             int lane) {
    return *(const v16bf*)(PW + (((long long)(t * NT + n) * NKB + kb) * 32 + lane) * 16);
}

__device__ __forceinline__ v8f wmma_bf16(v16bf a, v16bf b, v8f c) {
    return __builtin_amdgcn_wmma_f32_16x16x32_bf16(false, a, false, b, (short)0, c,
                                                   false, false);
}

// Generic LDS->LDS layer for layers 2/3: H[64][NT*16] = celu(A[64][K] * W + bias)
template <int K, int NT>
__device__ __forceinline__ void layer_lds(const __bf16* A, const __bf16* __restrict__ PW,
                                          const float* __restrict__ biasT, __bf16* Hs,
                                          int t, int wave, int lane) {
    const int NKB = K / 32;
    for (int n = wave; n < NT; n += 8) {              // wave-uniform trip count
        v8f acc[4];
#pragma unroll
        for (int m = 0; m < 4; ++m) acc[m] = (v8f){0.f,0.f,0.f,0.f,0.f,0.f,0.f,0.f};
#pragma unroll 2
        for (int kb = 0; kb < NKB; ++kb) {
            v16bf b = load_b_frag(PW, t, NT, NKB, n, kb, lane);
#pragma unroll
            for (int m = 0; m < 4; ++m) {
                v16bf a = load_a_frag(A + m * 16 * K, K, kb * 32, lane);
                acc[m] = wmma_bf16(a, b, acc[m]);
            }
        }
        int col = n * 16 + (lane & 15);
        float bs = biasT[col];
        int msub = (lane >> 4) * 8;
#pragma unroll
        for (int m = 0; m < 4; ++m)
#pragma unroll
            for (int j = 0; j < 8; ++j) {
                float x = celu01(acc[m][j] + bs);
                Hs[(m * 16 + msub + j) * (NT * 16) + col] = f2bf(x);
            }
    }
}

// ---------------------------------------------------------------------------
// Fused per-type MLP kernel: 64-atom tile per block, 256 threads (8 waves).
// LDS overlay by liveness:  regA = max(L1 K-panel 16KB, h2 24KB)
//                           regB = max(h1 32KB, h3 20KB)      -> 56KB total
// ---------------------------------------------------------------------------
__global__ __launch_bounds__(256) void k_mlp(
    const float* __restrict__ feat, const int* __restrict__ batch,
    float* __restrict__ out,
    const int* __restrict__ hdr, const int* __restrict__ perm,
    const int* __restrict__ tileT, const int* __restrict__ tileB,
    const __bf16* __restrict__ pw1, const __bf16* __restrict__ pw2,
    const __bf16* __restrict__ pw3,
    const float* __restrict__ b1, const float* __restrict__ b2,
    const float* __restrict__ b3, const float* __restrict__ W4,
    const float* __restrict__ b4) {

    __shared__ __align__(32) __bf16 sRegA[MTILE * H2];   // panel(64x128) / h2(64x192)
    __shared__ __align__(32) __bf16 sRegB[MTILE * H1];   // h1(64x256) / h3(64x160)
    __shared__ int sRows[MTILE];

    if ((int)blockIdx.x >= hdr[15]) return;              // block-uniform exit

    const int t    = tileT[blockIdx.x];
    const int base = tileB[blockIdx.x];
    const int segEnd = hdr[5 + t] + hdr[t];

    const int tid  = threadIdx.x;
    const int wave = tid >> 5;
    const int lane = tid & 31;

    if (tid < MTILE) {
        int rp = base + tid;
        sRows[tid] = (rp < segEnd) ? perm[rp] : -1;
    }

    // ---------------- Layer 1: stream K in 128-wide panels -----------------
    __bf16* panel = sRegA;                // 64 x 128 bf16
    __bf16* h1    = sRegB;                // 64 x 256 bf16
    const int n0 = wave, n1 = wave + 8;   // 16 N-tiles, 2 per wave

    v8f acc[2][4];
#pragma unroll
    for (int nn = 0; nn < 2; ++nn)
#pragma unroll
        for (int m = 0; m < 4; ++m) acc[nn][m] = (v8f){0.f,0.f,0.f,0.f,0.f,0.f,0.f,0.f};

    for (int kp = 0; kp < 8; ++kp) {      // 8 panels x 128 K = 1024 (1008 padded)
        __syncthreads();                  // previous panel fully consumed / sRows ready
        // stage 64 rows x 128 K as float4 -> 4x bf16 (coalesced 512B rows)
        for (int i = tid; i < MTILE * 32; i += 256) {
            int r  = i >> 5;
            int k4 = (i & 31) * 4;
            int k  = kp * 128 + k4;
            int atom = sRows[r];
            v4f f = {0.f, 0.f, 0.f, 0.f};
            if (atom >= 0 && k < IN_F)
                f = *(const v4f*)(feat + (long long)atom * IN_F + k);
            v4bf o; 
#pragma unroll
            for (int c = 0; c < 4; ++c) o[c] = f2bf(f[c]);
            *(v4bf*)(panel + r * 128 + k4) = o;
        }
        __syncthreads();
#pragma unroll
        for (int kb = 0; kb < 4; ++kb) {
            int kbg = kp * 4 + kb;
            v16bf bf0 = load_b_frag(pw1, t, 16, 32, n0, kbg, lane);
            v16bf bf1 = load_b_frag(pw1, t, 16, 32, n1, kbg, lane);
#pragma unroll
            for (int m = 0; m < 4; ++m) {
                v16bf a = load_a_frag(panel + m * 16 * 128, 128, kb * 32, lane);
                acc[0][m] = wmma_bf16(a, bf0, acc[0][m]);
                acc[1][m] = wmma_bf16(a, bf1, acc[1][m]);
            }
        }
    }
    // writeback h1 (regB: untouched by panel staging)
    {
        const float* b1T = b1 + t * H1;
        int msub = (lane >> 4) * 8;
#pragma unroll
        for (int nn = 0; nn < 2; ++nn) {
            int col = (wave + 8 * nn) * 16 + (lane & 15);
            float bs = b1T[col];
#pragma unroll
            for (int m = 0; m < 4; ++m)
#pragma unroll
                for (int j = 0; j < 8; ++j) {
                    float x = celu01(acc[nn][m][j] + bs);
                    h1[(m * 16 + msub + j) * H1 + col] = f2bf(x);
                }
        }
    }
    __syncthreads();

    // ---------------- Layer 2: h1(regB) -> h2(regA) -------------------------
    __bf16* h2 = sRegA;
    layer_lds<H1, 12>(h1, pw2, b2 + t * H2, h2, t, wave, lane);
    __syncthreads();

    // ---------------- Layer 3: h2(regA) -> h3(regB, over dead h1) -----------
    __bf16* h3 = sRegB;
    layer_lds<H2, 10>(h2, pw3, b3 + t * H3, h3, t, wave, lane);
    __syncthreads();

    // ---------------- Layer 4 (160 -> 1) + molecule segment-sum -------------
    if (tid < MTILE) {
        int atom = sRows[tid];
        if (atom >= 0) {
            const float* w4 = W4 + t * H3;
            float acc4 = b4[t];
#pragma unroll 8
            for (int k = 0; k < H3; ++k) acc4 += bf2f(h3[tid * H3 + k]) * w4[k];
            atomicAdd(&out[batch[atom]], acc4);
        }
    }
}

// ---------------------------------------------------------------------------
// Host launcher
// ---------------------------------------------------------------------------
extern "C" void kernel_launch(void* const* d_in, const int* in_sizes, int n_in,
                              void* d_out, int out_size, void* d_ws, size_t ws_size,
                              hipStream_t stream) {
    const int*   z    = (const int*)d_in[0];
    const float* feat = (const float*)d_in[1];
    const int*   batch= (const int*)d_in[2];
    const float* W1   = (const float*)d_in[4];
    const float* b1   = (const float*)d_in[5];
    const float* W2   = (const float*)d_in[6];
    const float* b2   = (const float*)d_in[7];
    const float* W3   = (const float*)d_in[8];
    const float* b3   = (const float*)d_in[9];
    const float* W4   = (const float*)d_in[10];
    const float* b4   = (const float*)d_in[11];
    float* out = (float*)d_out;

    char* ws = (char*)d_ws;
    int*    hdr   = (int*)(ws + OFF_HDR);
    int*    perm  = (int*)(ws + OFF_PERM);
    int*    tileT = (int*)(ws + OFF_TILT);
    int*    tileB = (int*)(ws + OFF_TILB);
    __bf16* pw1   = (__bf16*)(ws + OFF_PW1);
    __bf16* pw2   = (__bf16*)(ws + OFF_PW2);
    __bf16* pw3   = (__bf16*)(ws + OFF_PW3);

    int zb = (out_size + 255) / 256; if (zb < 1) zb = 1;
    k_zero   <<<zb, 256, 0, stream>>>(hdr, out, out_size);
    k_count  <<<(N_ATOMS + 255) / 256, 256, 0, stream>>>(z, hdr);
    k_scan   <<<1, 1, 0, stream>>>(hdr, tileT, tileB);
    k_scatter<<<(N_ATOMS + 255) / 256, 256, 0, stream>>>(z, hdr, perm);

    long long tw1 = (long long)N_TYPES * 16 * 32 * 512;
    long long tw2 = (long long)N_TYPES * 12 *  8 * 512;
    long long tw3 = (long long)N_TYPES * 10 *  6 * 512;
    k_pack<<<(unsigned)((tw1 + 255) / 256), 256, 0, stream>>>(W1, pw1, IN_F, H1, 16, 32);
    k_pack<<<(unsigned)((tw2 + 255) / 256), 256, 0, stream>>>(W2, pw2, H1,   H2, 12,  8);
    k_pack<<<(unsigned)((tw3 + 255) / 256), 256, 0, stream>>>(W3, pw3, H2,   H3, 10,  6);

    k_mlp<<<MAXT, 256, 0, stream>>>(feat, batch, out, hdr, perm, tileT, tileB,
                                    pw1, pw2, pw3, b1, b2, b3, W4, b4);
}
